// resnet_with_conv_regions_57990648430674
// MI455X (gfx1250) — matrix operand order repaired and checked by reading, not compiled, not run on hardware
//
#include <hip/hip_runtime.h>

// Problem constants (from the reference)
#define POOLSZ 14
#define HW     196          // 14*14
#define HWPAD  224          // 7 chunks of K=32
#define NBOX   64
#define CCH    2048
#define BB     64

typedef __attribute__((ext_vector_type(16))) _Float16 v16h;
typedef __attribute__((ext_vector_type(8)))  float    v8f;

// ---------------------------------------------------------------------------
// Kernel 1: fc_feats[b,c] = mean over 196 pixels. One wave32 per (b,c).
// ---------------------------------------------------------------------------
__global__ void fc_mean_kernel(const float* __restrict__ x,
                               float* __restrict__ fc) {
  const int wid  = (blockIdx.x * blockDim.x + threadIdx.x) >> 5;  // (b*C + c)
  const int lane = threadIdx.x & 31;
  const float* row = x + (size_t)wid * HW;
  float s = 0.0f;
  for (int hw = lane; hw < HW; hw += 32) s += row[hw];
  // wave32 tree reduction
  for (int o = 16; o > 0; o >>= 1) s += __shfl_down(s, o, 32);
  if (lane == 0) fc[wid] = s * (1.0f / (float)HW);
}

// ---------------------------------------------------------------------------
// Kernel 2: ROI masked-mean as per-batch GEMM on the f16 WMMA pipe.
//   grid = (C/16/8, NBOX/16, B); block = 256 threads = 8 waves.
//   Wave w computes the 16x16 tile D[ntile*16 .. +15, ctile*16 .. +15]
//   with ctile = blockIdx.x*8 + w, K = 224 (196 real + zero pad).
// ---------------------------------------------------------------------------
__global__ void roi_wmma_kernel(const float* __restrict__ x,
                                const float* __restrict__ boxes,
                                float* __restrict__ att) {
  __shared__ _Float16 smask[16][HWPAD + 8];  // +8 halves: LDS bank stagger
  __shared__ int      scoord[16][4];         // x1,y1,x2,y2 per box
  __shared__ float    sinv_area[16];

  const int b     = blockIdx.z;
  const int ntile = blockIdx.y;          // 0..3
  const int tid   = threadIdx.x;
  const int wave  = tid >> 5;            // 0..7
  const int lane  = tid & 31;

  // --- ROI coords: round-half-to-even (rintf == jnp.round) + degenerate fixup
  if (tid < 16) {
    const int n = ntile * 16 + tid;
    const float* bx = boxes + ((size_t)b * NBOX + n) * 4;
    int x1 = (int)rintf(bx[0] * (float)POOLSZ);
    int y1 = (int)rintf(bx[1] * (float)POOLSZ);
    int x2 = (int)rintf(bx[2] * (float)POOLSZ);
    int y2 = (int)rintf(bx[3] * (float)POOLSZ);
    if (x1 == x2) { if (x2 < POOLSZ) x2 += 1; else if (x1 > 0) x1 -= 1; }
    if (y1 == y2) { if (y2 < POOLSZ) y2 += 1; else if (y1 > 0) y1 -= 1; }
    scoord[tid][0] = x1; scoord[tid][1] = y1;
    scoord[tid][2] = x2; scoord[tid][3] = y2;
    sinv_area[tid] = 1.0f / (float)((y2 - y1) * (x2 - x1));
  }
  __syncthreads();

  // --- Build the binary mask tile in LDS: 16 boxes x 224 (padded) pixels, f16.
  for (int idx = tid; idx < 16 * HWPAD; idx += 256) {
    const int m  = idx / HWPAD;
    const int hw = idx - m * HWPAD;
    float v = 0.0f;
    if (hw < HW) {
      const int h = hw / POOLSZ, w = hw - h * POOLSZ;
      if (h >= scoord[m][1] && h < scoord[m][3] &&
          w >= scoord[m][0] && w < scoord[m][2]) v = 1.0f;
    }
    smask[m][hw] = (_Float16)v;
  }
  __syncthreads();

  // --- Per-wave 16x16 tile via v_wmma_f32_16x16x32_f16, K = 7 x 32.
  const int half_sel = lane >> 4;        // 0: lanes 0-15, 1: lanes 16-31
  const int m        = lane & 15;        // A row (box within tile)
  const int c_local  = lane & 15;        // B/D column (channel within tile)
  const int ctile    = blockIdx.x * 8 + wave;
  const int c        = ctile * 16 + c_local;

  const float* __restrict__ xr = x + ((size_t)b * CCH + c) * HW;
  __builtin_prefetch(xr, 0, 0);          // global_prefetch_b8: warm L2 path

  v8f acc = {};
#pragma unroll
  for (int k = 0; k < 7; ++k) {
    const int kbase = k * 32;

    // A fragment (mask): contiguous k order, same as B.
    //   lanes 0-15:  halves 0-15 -> K=kbase+0..15
    //   lanes 16-31: halves 0-15 -> K=kbase+16..31
    v16h afrag;
#pragma unroll
    for (int j = 0; j < 8; ++j) {
      afrag[j]     = smask[m][kbase + half_sel * 16 + j];
      afrag[j + 8] = smask[m][kbase + half_sel * 16 + 8 + j];
    }

    // B fragment (x, f32 -> f16): contiguous k order, same as A.
    //   lane holds column c_local; lanes 0-15 carry K=kbase+0..15,
    //   lanes 16-31 carry K=kbase+16..31 (16 consecutive pixels -> coalesced 64B).
    const int bbase = kbase + half_sel * 16;
    v16h bfrag;
#pragma unroll
    for (int j = 0; j < 16; ++j) {
      const int hw = bbase + j;
      const float v = (hw < HW) ? xr[hw] : 0.0f;
      bfrag[j] = (_Float16)v;
    }

    // D = A x B + C  (f32 accumulate; mask exact in f16)
    acc = __builtin_amdgcn_wmma_f32_16x16x32_f16(
        /*neg_a=*/false, afrag, /*neg_b=*/false, bfrag,
        /*c_mod=*/(short)0, acc, /*reuse_a=*/false, /*reuse_b=*/false);
  }

  // --- Scale by 1/area and store. D layout: VGPR r -> row (half_sel*8 + r),
  //     column c_local. att[b, n, c] with n = ntile*16 + row.
  float* __restrict__ outp =
      att + (((size_t)b * NBOX + (size_t)ntile * 16) * CCH) + c;
#pragma unroll
  for (int r = 0; r < 8; ++r) {
    const int row = half_sel * 8 + r;
    outp[(size_t)row * CCH] = acc[r] * sinv_area[row];
  }
}

// ---------------------------------------------------------------------------
extern "C" void kernel_launch(void* const* d_in, const int* in_sizes, int n_in,
                              void* d_out, int out_size, void* d_ws, size_t ws_size,
                              hipStream_t stream) {
  (void)in_sizes; (void)n_in; (void)out_size; (void)d_ws; (void)ws_size;
  const float* x     = (const float*)d_in[0];   // [B, C, 14, 14] f32
  const float* boxes = (const float*)d_in[1];   // [B, 64, 4] f32
  float* out = (float*)d_out;
  float* fc  = out;                             // [B, C]
  float* att = out + (size_t)BB * CCH;          // [B, N, C]

  // fc: one wave per (b,c) -> B*C waves -> B*C*32/256 blocks
  fc_mean_kernel<<<dim3((BB * CCH) / 8), 256, 0, stream>>>(x, fc);

  // ROI GEMM: (128 c-tiles / 8 waves, 4 n-tiles, 64 batches)
  roi_wmma_kernel<<<dim3(CCH / 16 / 8, NBOX / 16, BB), 256, 0, stream>>>(x, boxes, att);
}
